// MultiheadSelfAttention_86887188398137
// MI455X (gfx1250) — compile-verified, matrix operand order checked by reading, not run on hardware
//
#include <hip/hip_runtime.h>
#include <hip/hip_bf16.h>

typedef __bf16 bf16_t;
typedef __attribute__((ext_vector_type(16))) __bf16 v16bf;
typedef __attribute__((ext_vector_type(8)))  float  v8f;
typedef __attribute__((ext_vector_type(4)))  int    v4i;

// Pointer-to-int4 in global (AS1) and LDS (AS3) address spaces, matching the
// clang builtin signature vV4i*1 V4i*3 Ii Ii for global_load_async_to_lds_b128.
typedef __attribute__((address_space(1))) v4i* gas_v4i_p;
typedef __attribute__((address_space(3))) v4i* las_v4i_p;

#define NTOK    4096
#define DMODEL  2048
#define NHEAD   16
#define HEADDIM 128

// ---------------------------------------------------------------------------
// Async global->LDS copy helpers (gfx1250 GLOBAL_LOAD_ASYNC_TO_LDS_B128,
// tracked by ASYNCcnt). Guarded so compilation succeeds on toolchains that
// lack the builtin (falls back to a synchronous copy through VGPRs).
// ---------------------------------------------------------------------------
#if defined(__has_builtin)
#if __has_builtin(__builtin_amdgcn_global_load_async_to_lds_b128)
#define HAS_ASYNC_LDS 1
#endif
#endif
#ifndef HAS_ASYNC_LDS
#define HAS_ASYNC_LDS 0
#endif

__device__ __forceinline__ void async_copy_b128(const bf16_t* g, bf16_t* l) {
#if HAS_ASYNC_LDS
  __builtin_amdgcn_global_load_async_to_lds_b128(
      (gas_v4i_p)(g), (las_v4i_p)(l), 0, 0);
#else
  *(uint4*)l = *(const uint4*)g;
#endif
}

__device__ __forceinline__ void async_wait0() {
#if HAS_ASYNC_LDS
#if __has_builtin(__builtin_amdgcn_s_wait_asynccnt)
  __builtin_amdgcn_s_wait_asynccnt(0);
#else
  asm volatile("s_wait_asynccnt 0x0" ::: "memory");
#endif
#endif
}

// ---------------------------------------------------------------------------
// fp32 -> bf16 conversion
// ---------------------------------------------------------------------------
__global__ void cvt_f32_bf16(const float* __restrict__ src,
                             bf16_t* __restrict__ dst, int n) {
  int i = blockIdx.x * blockDim.x + threadIdx.x;
  if (i < n) dst[i] = (bf16_t)src[i];
}

// ---------------------------------------------------------------------------
// Tiled NT GEMM: C[M,Ncols] = A[M,K] * B[Ncols,K]^T, then (acc+bias)*scale.
// A, B bf16 row-major. 256 threads = 8 waves; block tile 128x128;
// wave w owns rows [w*16, w*16+16) x all 128 cols (8 wmma accumulators).
// Double-buffered LDS tiles filled with async global->LDS copies.
// ---------------------------------------------------------------------------
template <bool OUT_F32>
__global__ void __launch_bounds__(256)
gemm_nt(const bf16_t* __restrict__ A, const bf16_t* __restrict__ B,
        const float* __restrict__ bias, void* __restrict__ Cout,
        int M, int Ncols, int K, float out_scale) {
  __shared__ __attribute__((aligned(32))) bf16_t As[2][128][32];
  __shared__ __attribute__((aligned(32))) bf16_t Bs[2][128][32];

  const int t    = threadIdx.x;
  const int lane = t & 31;
  const int wave = t >> 5;
  const int bm   = blockIdx.y * 128;
  const int bn   = blockIdx.x * 128;

  v8f acc[8] = {};

  const int lrow  = t >> 1;   // 0..127
  const int lhalf = t & 1;    // which 16-element half of the 32-wide K slab
  const bf16_t* aLoad = A + (size_t)(bm + lrow) * K + lhalf * 16;
  const bf16_t* bLoad = B + (size_t)(bn + lrow) * K + lhalf * 16;

  auto stage = [&](int buf, int k0) {
    async_copy_b128(aLoad + k0,     &As[buf][lrow][lhalf * 16]);
    async_copy_b128(aLoad + k0 + 8, &As[buf][lrow][lhalf * 16 + 8]);
    async_copy_b128(bLoad + k0,     &Bs[buf][lrow][lhalf * 16]);
    async_copy_b128(bLoad + k0 + 8, &Bs[buf][lrow][lhalf * 16 + 8]);
  };

  // Prologue: fill buffer 0.
  stage(0, 0);
  async_wait0();
  __syncthreads();

  int buf = 0;
  for (int k0 = 0; k0 < K; k0 += 32) {
    const int nxt = k0 + 32;
    if (nxt < K) {
      stage(buf ^ 1, nxt);                       // async fill of other buffer
      if (nxt + 32 < K) {
        __builtin_prefetch(aLoad + nxt + 32, 0, 1);
        __builtin_prefetch(bLoad + nxt + 32, 0, 1);
      }
    }

    v16bf afrag =
        *(const v16bf*)&As[buf][wave * 16 + (lane & 15)][(lane >> 4) * 16];
#pragma unroll
    for (int n = 0; n < 8; ++n) {
      v16bf bfrag =
          *(const v16bf*)&Bs[buf][n * 16 + (lane & 15)][(lane >> 4) * 16];
      acc[n] = __builtin_amdgcn_wmma_f32_16x16x32_bf16(
          false, afrag, false, bfrag, (short)0, acc[n], false, false);
    }

    async_wait0();      // our async fills of buf^1 are done
    __syncthreads();    // everyone done reading buf & filling buf^1
    buf ^= 1;
  }

  // Epilogue: C-layout element (m,n): lane = n + 16*(m/8), vgpr i = m%8.
#pragma unroll
  for (int n = 0; n < 8; ++n) {
    const int gn = bn + n * 16 + (lane & 15);
    const float bv = bias[gn];
#pragma unroll
    for (int i = 0; i < 8; ++i) {
      const int gm = bm + wave * 16 + i + 8 * (lane >> 4);
      float v = (acc[n][i] + bv) * out_scale;
      if (OUT_F32)
        ((float*)Cout)[(size_t)gm * Ncols + gn] = v;
      else
        ((bf16_t*)Cout)[(size_t)gm * Ncols + gn] = (bf16_t)v;
    }
  }
}

// ---------------------------------------------------------------------------
// Flash attention forward. Grid: (NTOK/128, NHEAD), 256 threads (8 waves).
// Wave w handles query rows [blk*128 + w*16 .. +16). Q pre-scaled by 1/sqrt(hd).
// Streams 64-wide K/V column blocks through LDS; K tile filled with async
// global->LDS copies, V stored transposed so P*V is also an NT contraction.
// ---------------------------------------------------------------------------
__global__ void __launch_bounds__(256)
attn_fwd(const bf16_t* __restrict__ Q, const bf16_t* __restrict__ Kmat,
         const bf16_t* __restrict__ V, bf16_t* __restrict__ O) {
  __shared__ __attribute__((aligned(32))) bf16_t Ks[64][128];      // 16 KB
  __shared__ __attribute__((aligned(32))) bf16_t Vt[128][64];      // 16 KB
  __shared__ __attribute__((aligned(32))) bf16_t Ps[8][16][64];    // 16 KB

  const int t    = threadIdx.x;
  const int lane = t & 31;
  const int wave = t >> 5;
  const int rowBase = blockIdx.x * 128;
  const int h       = blockIdx.y;
  const size_t hoff = (size_t)h * HEADDIM;

  // Q fragments (16 rows x 128 K) held in registers for the whole kernel.
  v16bf qa[4];
  {
    const size_t r = (size_t)(rowBase + wave * 16 + (lane & 15));
    const bf16_t* qp = Q + r * DMODEL + hoff + (lane >> 4) * 16;
#pragma unroll
    for (int kc = 0; kc < 4; ++kc) qa[kc] = *(const v16bf*)(qp + kc * 32);
  }

  float rmax[8], rsum[8];
#pragma unroll
  for (int i = 0; i < 8; ++i) { rmax[i] = -__builtin_inff(); rsum[i] = 0.f; }
  v8f cacc[8] = {};

  const int lrow = t >> 2;  // 0..63 : row within 64-row K/V block
  const int lq   = t & 3;   // quarter of the 128-wide head dim

  for (int cblk = 0; cblk < NTOK / 64; ++cblk) {
    // --- K block: async global->LDS straight copy (4 x b128 per thread) ---
    {
      const bf16_t* kp =
          Kmat + (size_t)(cblk * 64 + lrow) * DMODEL + hoff + lq * 32;
#pragma unroll
      for (int e = 0; e < 4; ++e)
        async_copy_b128(kp + e * 8, &Ks[lrow][lq * 32 + e * 8]);
    }
    // --- V block: load through VGPRs and transpose into Vt[hd][col] ---
    {
      const bf16_t* vp =
          V + (size_t)(cblk * 64 + lrow) * DMODEL + hoff + lq * 32;
      bf16_t tmp[32];
      {
        const uint4* s = (const uint4*)vp;
        uint4* d = (uint4*)tmp;
        d[0] = s[0]; d[1] = s[1]; d[2] = s[2]; d[3] = s[3];
      }
#pragma unroll
      for (int e = 0; e < 32; ++e) Vt[lq * 32 + e][lrow] = tmp[e];
    }
    async_wait0();
    __syncthreads();

    // --- S = Q * K^T : 16 x 64 per wave, f32 accum ---
    v8f sacc[4] = {};
#pragma unroll
    for (int n = 0; n < 4; ++n) {
#pragma unroll
      for (int kc = 0; kc < 4; ++kc) {
        v16bf kf =
            *(const v16bf*)&Ks[n * 16 + (lane & 15)][kc * 32 + (lane >> 4) * 16];
        sacc[n] = __builtin_amdgcn_wmma_f32_16x16x32_bf16(
            false, qa[kc], false, kf, (short)0, sacc[n], false, false);
      }
    }

    // --- online softmax: row m lives in the 16 lanes of one half-wave ---
#pragma unroll
    for (int i = 0; i < 8; ++i) {
      float bmax = sacc[0][i];
#pragma unroll
      for (int n = 1; n < 4; ++n) bmax = fmaxf(bmax, sacc[n][i]);
#pragma unroll
      for (int m = 1; m < 16; m <<= 1)
        bmax = fmaxf(bmax, __shfl_xor(bmax, m, 32));
      const float nm = fmaxf(rmax[i], bmax);
      const float alpha = __expf(rmax[i] - nm);
      rmax[i] = nm;
      float ls = 0.f;
#pragma unroll
      for (int n = 0; n < 4; ++n) {
        float p = __expf(sacc[n][i] - nm);
        sacc[n][i] = p;
        ls += p;
      }
#pragma unroll
      for (int m = 1; m < 16; m <<= 1) ls += __shfl_xor(ls, m, 32);
      rsum[i] = rsum[i] * alpha + ls;
#pragma unroll
      for (int tt = 0; tt < 8; ++tt) cacc[tt][i] *= alpha;
    }

    // --- restage P through per-wave LDS to get a WMMA-A fragment layout ---
#pragma unroll
    for (int n = 0; n < 4; ++n)
#pragma unroll
      for (int i = 0; i < 8; ++i)
        Ps[wave][i + 8 * (lane >> 4)][n * 16 + (lane & 15)] =
            (bf16_t)sacc[n][i];

    asm volatile("s_wait_dscnt 0" ::: "memory");  // own-wave LDS RAW

    // --- ctx += P * V  (K-dim = 64 columns of this block) ---
#pragma unroll
    for (int tt = 0; tt < 8; ++tt) {
#pragma unroll
      for (int kc = 0; kc < 2; ++kc) {
        v16bf pf =
            *(const v16bf*)&Ps[wave][lane & 15][kc * 32 + (lane >> 4) * 16];
        v16bf vf =
            *(const v16bf*)&Vt[tt * 16 + (lane & 15)][kc * 32 + (lane >> 4) * 16];
        cacc[tt] = __builtin_amdgcn_wmma_f32_16x16x32_bf16(
            false, pf, false, vf, (short)0, cacc[tt], false, false);
      }
    }
    __syncthreads();
  }

  // --- normalize and store ctx (bf16, [N, D] with head offset) ---
#pragma unroll
  for (int tt = 0; tt < 8; ++tt) {
#pragma unroll
    for (int i = 0; i < 8; ++i) {
      const size_t gr = (size_t)(rowBase + wave * 16 + i + 8 * (lane >> 4));
      const size_t gc = hoff + tt * 16 + (lane & 15);
      O[gr * DMODEL + gc] = (bf16_t)(cacc[tt][i] / rsum[i]);
    }
  }
}

// ---------------------------------------------------------------------------
// Host-side launcher
// ---------------------------------------------------------------------------
extern "C" void kernel_launch(void* const* d_in, const int* in_sizes, int n_in,
                              void* d_out, int out_size, void* d_ws,
                              size_t ws_size, hipStream_t stream) {
  (void)in_sizes; (void)n_in; (void)out_size; (void)ws_size;

  const float* x  = (const float*)d_in[0];
  const float* Wq = (const float*)d_in[1];
  const float* bq = (const float*)d_in[2];
  const float* Wk = (const float*)d_in[3];
  const float* bk = (const float*)d_in[4];
  const float* Wv = (const float*)d_in[5];
  const float* bv = (const float*)d_in[6];
  const float* Wo = (const float*)d_in[7];
  const float* bo = (const float*)d_in[8];

  char* ws = (char*)d_ws;
  const size_t ND = (size_t)NTOK * DMODEL;    // 8,388,608
  const size_t DD = (size_t)DMODEL * DMODEL;  // 4,194,304

  bf16_t* xb   = (bf16_t*)(ws);
  bf16_t* wqb  = (bf16_t*)(ws + 2 * ND);
  bf16_t* wkb  = (bf16_t*)(ws + 2 * ND + 2 * DD);
  bf16_t* wvb  = (bf16_t*)(ws + 2 * ND + 4 * DD);
  bf16_t* wob  = (bf16_t*)(ws + 2 * ND + 6 * DD);
  bf16_t* qb   = (bf16_t*)(ws + 2 * ND + 8 * DD);
  bf16_t* kb   = (bf16_t*)(ws + 4 * ND + 8 * DD);
  bf16_t* vbuf = (bf16_t*)(ws + 6 * ND + 8 * DD);
  bf16_t* ctxb = (bf16_t*)(ws + 8 * ND + 8 * DD);

  const int thr = 256;
  cvt_f32_bf16<<<(int)((ND + thr - 1) / thr), thr, 0, stream>>>(x, xb, (int)ND);
  cvt_f32_bf16<<<(int)((DD + thr - 1) / thr), thr, 0, stream>>>(Wq, wqb, (int)DD);
  cvt_f32_bf16<<<(int)((DD + thr - 1) / thr), thr, 0, stream>>>(Wk, wkb, (int)DD);
  cvt_f32_bf16<<<(int)((DD + thr - 1) / thr), thr, 0, stream>>>(Wv, wvb, (int)DD);
  cvt_f32_bf16<<<(int)((DD + thr - 1) / thr), thr, 0, stream>>>(Wo, wob, (int)DD);

  dim3 gg(DMODEL / 128, NTOK / 128);  // (cols, rows)
  const float qscale = 0.08838834764831845f;  // 1/sqrt(128), folded into q
  gemm_nt<false><<<gg, 256, 0, stream>>>(xb, wqb, bq, qb, NTOK, DMODEL, DMODEL, qscale);
  gemm_nt<false><<<gg, 256, 0, stream>>>(xb, wkb, bk, kb, NTOK, DMODEL, DMODEL, 1.0f);
  gemm_nt<false><<<gg, 256, 0, stream>>>(xb, wvb, bv, vbuf, NTOK, DMODEL, DMODEL, 1.0f);

  dim3 ga(NTOK / 128, NHEAD);
  attn_fwd<<<ga, 256, 0, stream>>>(qb, kb, vbuf, ctxb);

  gemm_nt<true><<<gg, 256, 0, stream>>>(ctxb, wob, bo, d_out, NTOK, DMODEL, DMODEL, 1.0f);
}
